// LorentzianKAN_15693810499706
// MI455X (gfx1250) — matrix operand order repaired and checked by reading, not compiled, not run on hardware
//
#include <hip/hip_runtime.h>
#include <hip/hip_bf16.h>
#include <stdint.h>

// ---------------------------------------------------------------------------
// CDNA5 / gfx1250 types for v_wmma_f32_16x16x32_bf16
// ---------------------------------------------------------------------------
typedef __attribute__((ext_vector_type(16))) __bf16 v16bf;
typedef __attribute__((ext_vector_type(8)))  float  v8f;
typedef __attribute__((ext_vector_type(4)))  int    v4i;
typedef __attribute__((ext_vector_type(2)))  float  v2f;
typedef __attribute__((ext_vector_type(2)))  __bf16 v2bf;

typedef __attribute__((address_space(1))) v4i* glob_v4i_ptr;
typedef __attribute__((address_space(3))) v4i* lds_v4i_ptr;

union FragB16 {
    v16bf v;
    unsigned short s[16];
    unsigned int   w[8];
    uint4          q[2];
};

__device__ __forceinline__ unsigned short f2bf_bits(float f) {
    union { float f; unsigned u; } c; c.f = f;
    unsigned u = c.u;
    unsigned r = u + 0x7FFFu + ((u >> 16) & 1u);   // round-to-nearest-even
    return (unsigned short)(r >> 16);
}

// Pack two f32 -> packed bf16x2 (RNE).
__device__ __forceinline__ unsigned pack2bf(float a, float b) {
#if __has_builtin(__builtin_amdgcn_cvt_pk_bf16_f32)
    union { v2bf v; unsigned u; } c;
    c.v = __builtin_amdgcn_cvt_pk_bf16_f32(a, b);
    return c.u;
#else
    // fptrunc <2 x float> -> <2 x bfloat>; backend selects packed HW cvt when
    // available on gfx1250, otherwise legalizes to the integer sequence.
    v2f f; f.x = a; f.y = b;
    union { v2bf v; unsigned u; } c;
    c.v = __builtin_convertvector(f, v2bf);
    return c.u;
#endif
}

// Async global->LDS staging (gfx1250 GLOBAL_LOAD_ASYNC_TO_LDS_B128), guarded.
#if defined(__gfx1250__) && \
    __has_builtin(__builtin_amdgcn_global_load_async_to_lds_b128) && \
    __has_builtin(__builtin_amdgcn_s_wait_asynccnt)
#define USE_ASYNC_LDS 1
#else
#define USE_ASYNC_LDS 0
#endif

// ---------------------------------------------------------------------------
// f32 -> bf16 weight conversion (one-time per launch, tiny)
// ---------------------------------------------------------------------------
__global__ __launch_bounds__(256) void cvt_f32_bf16(const float* __restrict__ in,
                                                    unsigned short* __restrict__ out,
                                                    int n) {
    int i = blockIdx.x * blockDim.x + threadIdx.x;
    if (i < n) out[i] = f2bf_bits(in[i]);
}

// ---------------------------------------------------------------------------
// Fused KAN layer:
//   out[b,o] = base_b[o] + sum_k silu(x[b,k])*Wb[o,k]
//            + sum_{i,g} lorentz(LN(x)[b,i], grid[g]) * Ws[o,i,g]
// Block = 128 threads (4 waves), owns one 16-row tile of the batch.
// The spline contraction is a [16 x K] @ [K x DOUT] GEMM on bf16 WMMA with
// K = DIN*16, processed in NSPLIT K-chunks so the LDS basis tile stays small
// (33 KB) and 6-7 blocks fit per 320 KB WGP for latency hiding.
// ---------------------------------------------------------------------------
template<int DIN, int DOUT, int NCT, int NSPLIT>  // 4*NCT*16 == DOUT
__global__ __launch_bounds__(128) void kan_layer(
    const float* __restrict__ x,                 // [B, DIN]
    const float* __restrict__ ln_w,
    const float* __restrict__ ln_b,
    const unsigned short* __restrict__ wbase,    // [DOUT, DIN]   bf16
    const float* __restrict__ bbase,             // [DOUT]
    const unsigned short* __restrict__ wspl,     // [DOUT, DIN*16] bf16
    float* __restrict__ out)                     // [B, DOUT]
{
    constexpr int K  = DIN * 16;      // full spline contraction depth
    constexpr int KS = K / NSPLIT;    // K per chunk (1024 for both layers)
    constexpr int IS = DIN / NSPLIT;  // input dims per chunk
    constexpr int XS = DIN + 4;       // padded f32 LDS row stride (bank spread)
    constexpr int BS = KS + 8;        // padded bf16 LDS row stride (16B aligned)

    __shared__ float          lds_x[16 * XS];
    __shared__ unsigned short lds_bas[16 * BS];
    __shared__ float          lds_ps[128], lds_ps2[128];
    __shared__ float          lds_mu[16], lds_rs[16];

    const int t  = threadIdx.x;
    const int b0 = blockIdx.x * 16;

    // ---- stage 16 x DIN tile of x into LDS (async DMA to LDS) ----
    {
        const float4* g = (const float4*)(x + (size_t)b0 * DIN);
        #pragma unroll
        for (int i = 0; i < DIN / 32; ++i) {     // 16*DIN/4 float4 over 128 threads
            int f   = t + i * 128;
            int row = (f * 4) / DIN;
            int k   = (f * 4) % DIN;
#if USE_ASYNC_LDS
            __builtin_amdgcn_global_load_async_to_lds_b128(
                (glob_v4i_ptr)(g + f),
                (lds_v4i_ptr)&lds_x[row * XS + k],
                0, 0);
#else
            *(float4*)&lds_x[row * XS + k] = g[f];
#endif
        }
#if USE_ASYNC_LDS
        __builtin_amdgcn_s_wait_asynccnt(0);
#endif
    }
    __syncthreads();

    // ---- layernorm stats: 8 segments x 16 rows ----
    {
        int row = t & 15, seg = t >> 4;
        float s = 0.f, s2 = 0.f;
        const float* xr = &lds_x[row * XS + seg * (DIN / 8)];
        #pragma unroll 4
        for (int k = 0; k < DIN / 8; ++k) { float v = xr[k]; s += v; s2 += v * v; }
        lds_ps[t]  = s;
        lds_ps2[t] = s2;
    }
    __syncthreads();
    if (t < 16) {
        float s = 0.f, s2 = 0.f;
        #pragma unroll
        for (int seg = 0; seg < 8; ++seg) { s += lds_ps[seg * 16 + t]; s2 += lds_ps2[seg * 16 + t]; }
        float mu  = s * (1.0f / DIN);
        float var = s2 * (1.0f / DIN) - mu * mu;
        lds_mu[t] = mu;
        lds_rs[t] = rsqrtf(var + 1e-5f);
    }
    __syncthreads();

    // ---- WMMA tiling roles ----
    const int wave = t >> 5;
    const int lane = t & 31;
    const int col  = lane & 15;     // C/D: column = lane%16
    const int hi   = lane >> 4;     // C/D: rows r+8*hi ; A/B: K-half select
    const int row  = col;           // A fragment row for this lane

    v8f acc[NCT];
    #pragma unroll
    for (int ct = 0; ct < NCT; ++ct) {
        float bb = bbase[(wave * NCT + ct) * 16 + col];
        #pragma unroll
        for (int r = 0; r < 8; ++r) acc[ct][r] = bb;
    }

    // --- base branch: silu(x) @ Wb.T, K = DIN ---
    #pragma unroll
    for (int kb = 0; kb < DIN / 32; ++kb) {
        // 16-bit A 16x32 layout: lane holds K = {kb*32+8*hi+0..7, +16..+23}
        const float* xp = &lds_x[row * XS + kb * 32 + 8 * hi];
        float4 xa = *(const float4*)(xp + 0);
        float4 xb = *(const float4*)(xp + 4);
        float4 xc = *(const float4*)(xp + 16);
        float4 xd = *(const float4*)(xp + 20);
        float xv[16] = { xa.x, xa.y, xa.z, xa.w, xb.x, xb.y, xb.z, xb.w,
                         xc.x, xc.y, xc.z, xc.w, xd.x, xd.y, xd.z, xd.w };
        FragB16 a;
        #pragma unroll
        for (int e = 0; e < 16; e += 2) {
            float v0 = xv[e],     s0 = v0 * __builtin_amdgcn_rcpf(1.0f + __expf(-v0));
            float v1 = xv[e + 1], s1 = v1 * __builtin_amdgcn_rcpf(1.0f + __expf(-v1));
            a.w[e >> 1] = pack2bf(s0, s1);
        }
        #pragma unroll
        for (int ct = 0; ct < NCT; ++ct) {
            int o = (wave * NCT + ct) * 16 + col;
            const uint4* bp = (const uint4*)(wbase + (size_t)o * DIN + kb * 32 + 8 * hi);
            FragB16 b; b.q[0] = bp[0]; b.q[1] = bp[2];
            acc[ct] = __builtin_amdgcn_wmma_f32_16x16x32_bf16(
                false, a.v, false, b.v, (short)0, acc[ct], false, false);
        }
    }

    // --- spline branch, NSPLIT K-chunks: build basis tile, then GEMM it ---
    for (int sp = 0; sp < NSPLIT; ++sp) {
        __syncthreads();   // previous chunk's GEMM done before overwriting basis
        {
            int brow = t >> 3;                           // 8 threads per row
            int i0   = (t & 7) * (IS / 8);
            float mu = lds_mu[brow], rs = lds_rs[brow];
            #pragma unroll 2
            for (int ii = 0; ii < IS / 8; ++ii) {
                int il = i0 + ii;                        // i within chunk
                int ig = sp * IS + il;                   // global input dim
                float xn = (lds_x[brow * XS + ig] - mu) * rs * ln_w[ig] + ln_b[ig];
                unsigned* dst = (unsigned*)&lds_bas[brow * BS + il * 16];
                #pragma unroll
                for (int gi = 0; gi < 16; gi += 2) {
                    float d0 = xn - (-1.0f + 0.2f * (float)gi);       // grid=linspace(-1,2,16)
                    float d1 = xn - (-1.0f + 0.2f * (float)(gi + 1));
                    float b0v = 0.04f * __builtin_amdgcn_rcpf(d0 * d0 + 0.04f);
                    float b1v = 0.04f * __builtin_amdgcn_rcpf(d1 * d1 + 0.04f);
                    dst[gi >> 1] = pack2bf(b0v, b1v);
                }
            }
        }
        __syncthreads();

        #pragma unroll 2
        for (int kb = 0; kb < KS / 32; ++kb) {
            FragB16 a;
            // e0..7  -> (i=2kb,   g=8*hi+0..7) : 16 contiguous bytes
            // e8..15 -> (i=2kb+1, g=8*hi+0..7) : 16 contiguous bytes
            a.q[0] = *(const uint4*)&lds_bas[row * BS + kb * 32 +      8 * hi];
            a.q[1] = *(const uint4*)&lds_bas[row * BS + kb * 32 + 16 + 8 * hi];
            #pragma unroll
            for (int ct = 0; ct < NCT; ++ct) {
                int o = (wave * NCT + ct) * 16 + col;
                const unsigned short* wp =
                    wspl + (size_t)o * K + sp * KS + kb * 32 + 8 * hi;
                if (ct == 0 && kb + 1 < KS / 32)
                    __builtin_prefetch(wp + 32, 0, 1);   // global_prefetch next K-block
                const uint4* bp = (const uint4*)wp;
                FragB16 b; b.q[0] = bp[0]; b.q[1] = bp[2];
                acc[ct] = __builtin_amdgcn_wmma_f32_16x16x32_bf16(
                    false, a.v, false, b.v, (short)0, acc[ct], false, false);
            }
        }
    }

    // ---- store: C/D layout -> VGPR r is (row b0+r+8*hi, col o) ----
    #pragma unroll
    for (int ct = 0; ct < NCT; ++ct) {
        int o = (wave * NCT + ct) * 16 + col;
        #pragma unroll
        for (int r = 0; r < 8; ++r)
            out[(size_t)(b0 + r + 8 * hi) * DOUT + o] = acc[ct][r];
    }
}

// ---------------------------------------------------------------------------
extern "C" void kernel_launch(void* const* d_in, const int* in_sizes, int n_in,
                              void* d_out, int out_size, void* d_ws, size_t ws_size,
                              hipStream_t stream) {
    (void)in_sizes; (void)n_in; (void)out_size; (void)ws_size;

    const float* x    = (const float*)d_in[0];
    const float* lnw1 = (const float*)d_in[1];
    const float* lnb1 = (const float*)d_in[2];
    const float* bw1  = (const float*)d_in[3];
    const float* bb1  = (const float*)d_in[4];
    const float* sw1  = (const float*)d_in[5];
    const float* lnw2 = (const float*)d_in[6];
    const float* lnb2 = (const float*)d_in[7];
    const float* bw2  = (const float*)d_in[8];
    const float* bb2  = (const float*)d_in[9];
    const float* sw2  = (const float*)d_in[10];

    const int B = 32768, d0 = 128, d1 = 256, d2 = 128, G = 16;

    // workspace layout: h[B,d1] f32, then bf16 weights
    char* ws = (char*)d_ws;
    float* h = (float*)ws;
    size_t off = (size_t)B * d1 * sizeof(float);
    unsigned short* bw1b = (unsigned short*)(ws + off); off += (size_t)d1 * d0 * 2;
    unsigned short* sw1b = (unsigned short*)(ws + off); off += (size_t)d1 * d0 * G * 2;
    unsigned short* bw2b = (unsigned short*)(ws + off); off += (size_t)d2 * d1 * 2;
    unsigned short* sw2b = (unsigned short*)(ws + off); off += (size_t)d2 * d1 * G * 2;

    auto cdiv = [](int a, int b) { return (a + b - 1) / b; };
    cvt_f32_bf16<<<cdiv(d1 * d0, 256),     256, 0, stream>>>(bw1, bw1b, d1 * d0);
    cvt_f32_bf16<<<cdiv(d1 * d0 * G, 256), 256, 0, stream>>>(sw1, sw1b, d1 * d0 * G);
    cvt_f32_bf16<<<cdiv(d2 * d1, 256),     256, 0, stream>>>(bw2, bw2b, d2 * d1);
    cvt_f32_bf16<<<cdiv(d2 * d1 * G, 256), 256, 0, stream>>>(sw2, sw2b, d2 * d1 * G);

    // layer 1: [B,128] -> [B,256], K=2048 in 2 chunks (4 waves x 4 col tiles)
    kan_layer<128, 256, 4, 2><<<B / 16, 128, 0, stream>>>(x, lnw1, lnb1, bw1b, bb1, sw1b, h);
    // layer 2: [B,256] -> [B,128], K=4096 in 4 chunks (4 waves x 2 col tiles)
    kan_layer<256, 128, 2, 4><<<B / 16, 128, 0, stream>>>(h, lnw2, lnb2, bw2b, bb2, sw2b, (float*)d_out);
}
